// SpatialConv2d_58772332478718
// MI455X (gfx1250) — matrix operand order repaired and checked by reading, not compiled, not run on hardware
//
#include <hip/hip_runtime.h>

typedef __attribute__((ext_vector_type(16))) _Float16 v16h;
typedef __attribute__((ext_vector_type(8)))  float    v8f;

// Problem constants (from reference setup_inputs)
constexpr int BB   = 16;          // batch
constexpr int CIN  = 64;          // conv input channels
constexpr int OCH  = 128;         // conv output channels
constexpr int XC   = 69;          // total x channels (64 feat + 5 shp)
constexpr int SHC  = 5;
constexpr int HH_  = 32, WW_ = 32;
constexpr int HP   = 34, WP  = 34;     // edge-padded
constexpr int PIX  = HH_ * WW_;        // 1024
constexpr int NPIX = BB * PIX;         // 16384  (GEMM N)
constexpr int KD   = CIN * 9;          // 576    (GEMM K)
constexpr int NS   = KD / 32;          // 18 k-slices of 32
constexpr int OT   = OCH / 16;         // 8 o-tiles of 16
constexpr int NT   = NPIX / 16;        // 1024 n-tiles of 16 pixels
constexpr int CHW  = HP * WP;          // 1156 (channel plane, padded)

// GEMM contraction uses the permuted index  k = t*64 + ci  (t = ki*3+kj tap).
// This makes every B-fragment gather offset a compile-time immediate.

// ---------------------------------------------------------------------------
// K1: pack conv weights (O,Cin,3,3) f32 -> WMMA A-fragment layout f16,
// using the permuted k ordering.
// A layout: lane<16: M=lane, K = hh ;  lanes>=16: K += 16.
// ---------------------------------------------------------------------------
__global__ void k_pack_w(const float* __restrict__ w, _Float16* __restrict__ apack) {
    int tid = blockIdx.x * blockDim.x + threadIdx.x;
    if (tid >= OT * NS * 32 * 16) return;
    int hh   = tid & 15;
    int lane = (tid >> 4) & 31;
    int s    = (tid >> 9) % NS;
    int ot   = tid / (NS * 512);
    int kl = s * 32 + hh + (lane >= 16 ? 16 : 0); // logical k
    int t  = kl >> 6;            // tap   (k = t*64 + ci)
    int ci = kl & 63;            // input channel
    int o  = ot * 16 + (lane & 15);
    apack[tid] = (_Float16)w[(o * CIN + ci) * 9 + t];
}

// ---------------------------------------------------------------------------
// K2: edge-pad. feat channels -> f16 featp (B,64,34,34); shp -> f32 shpp (B,5,34,34)
// ---------------------------------------------------------------------------
__global__ void k_pad(const float* __restrict__ x,
                      _Float16* __restrict__ featp, float* __restrict__ shpp) {
    int tid = blockIdx.x * blockDim.x + threadIdx.x;
    const int NF  = BB * CIN * CHW;
    const int NSH = BB * SHC * CHW;
    if (tid < NF) {
        int wp = tid % WP; int t = tid / WP;
        int hp = t % HP;   t /= HP;
        int ci = t % CIN;  int b = t / CIN;
        int h = hp - 1; h = h < 0 ? 0 : (h > HH_ - 1 ? HH_ - 1 : h);
        int w = wp - 1; w = w < 0 ? 0 : (w > WW_ - 1 ? WW_ - 1 : w);
        featp[tid] = (_Float16)x[((size_t)(b * XC + ci) * HH_ + h) * WW_ + w];
    } else if (tid < NF + NSH) {
        int i = tid - NF;
        int wp = i % WP; int t = i / WP;
        int hp = t % HP; t /= HP;
        int ch = t % SHC; int b = t / SHC;
        int h = hp - 1; h = h < 0 ? 0 : (h > HH_ - 1 ? HH_ - 1 : h);
        int w = wp - 1; w = w < 0 ? 0 : (w > WW_ - 1 ? WW_ - 1 : w);
        shpp[i] = x[((size_t)(b * XC + CIN + ch) * HH_ + h) * WW_ + w];
    }
}

// ---------------------------------------------------------------------------
// K3: implicit-GEMM conv via v_wmma_f32_16x16x32_f16.
// One wave owns a 16-pixel n-tile; 18 B fragments live in registers and are
// reused across all 8 o-tiles (144 WMMAs / wave).
// B layout: lane<16 -> N=lane, K=hh ; lane>=16 -> N=lane-16, K=hh+16.
// With k = t*64+ci, slice s has tap t=s>>1 and ci = (s&1)*32 + hh + 16*(lane>=16),
// so each gather is  fbase[ ((s&1)*32+hh)*1156 + ki*34 + kj ]  (immediate offset).
// ---------------------------------------------------------------------------
__global__ void __launch_bounds__(32)
k_conv_wmma(const _Float16* __restrict__ featp, const _Float16* __restrict__ apack,
            const float* __restrict__ bias, float* __restrict__ conv_out) {
    const int lane = threadIdx.x;
    const int n = blockIdx.x * 16 + (lane & 15);      // pixel id for this lane
    const int b = n >> 10, p = n & 1023, h = p >> 5, w = p & 31;
    const int hi = (lane >= 16) ? 1 : 0;              // upper half-wave?

    // one runtime base per lane; +16 channel planes for the upper half-wave
    const _Float16* fbase =
        featp + ((size_t)b * CIN + hi * 16) * CHW + h * WP + w;

    v16h bfr[NS];
#pragma unroll
    for (int s = 0; s < NS; ++s) {
        const int t  = s >> 1;            // compile-time tap
        const int ki = t / 3, kj = t % 3; // compile-time
        const int cb = (s & 1) * 32;      // compile-time channel base
#pragma unroll
        for (int hh = 0; hh < 16; ++hh) {
            bfr[s][hh] = fbase[(cb + hh) * CHW + ki * WP + kj];
        }
    }

    const int mo = hi * 8;
#pragma unroll 1
    for (int ot = 0; ot < OT; ++ot) {
        v8f acc = {};
        const _Float16* ap = apack + (size_t)ot * NS * 32 * 16;
#pragma unroll
        for (int s = 0; s < NS; ++s) {
            v16h afr = *(const v16h*)(ap + (s * 32 + lane) * 16);
            acc = __builtin_amdgcn_wmma_f32_16x16x32_f16(
                false, afr, false, bfr[s], (short)0, acc, false, false);
        }
        // D tile: lane -> N = lane&15, VGPR r -> M = r + 8*hi
        float* obase = conv_out + ((size_t)b * OCH + ot * 16 + mo) * PIX + p;
        const float* bb = bias + ot * 16 + mo;
#pragma unroll
        for (int r = 0; r < 8; ++r) {
            obase[r * PIX] = acc[r] + bb[r];
        }
    }
}

// ---------------------------------------------------------------------------
// K4: shape-difference: sdw[b,o,p] = sum_{ch,t} |window[b,ch,t,p] - sk[o,ch,t]|
// window = patches with ch 0,1 centered on t=4 (padded (h+1,w+1)).
// ---------------------------------------------------------------------------
__global__ void k_sdw(const float* __restrict__ shpp, const float* __restrict__ sk,
                      float* __restrict__ sdw) {
    int pix = blockIdx.x * blockDim.x + threadIdx.x;
    if (pix >= NPIX) return;
    int b = pix >> 10, p = pix & 1023, h = p >> 5, w = p & 31;
    const float* sb = shpp + (size_t)b * SHC * CHW;

    float win[SHC * 9];
#pragma unroll
    for (int ch = 0; ch < SHC; ++ch) {
        float cen = sb[(ch * HP + h + 1) * WP + (w + 1)];
#pragma unroll
        for (int t = 0; t < 9; ++t) {
            int ki = t / 3, kj = t % 3;
            float v = sb[(ch * HP + h + ki) * WP + (w + kj)];
            win[ch * 9 + t] = (ch < 2) ? (v - cen) : v;
        }
    }
    for (int o = 0; o < OCH; ++o) {
        const float* kk = sk + o * SHC * 9;
        float s = 0.f;
#pragma unroll
        for (int i = 0; i < SHC * 9; ++i) s += fabsf(win[i] - kk[i]);
        sdw[((size_t)b * OCH + o) * PIX + p] = s;
    }
}

// ---------------------------------------------------------------------------
// K5: per-channel mean / inv-std for conv_out (c<128) and sdw (c>=128).
// ---------------------------------------------------------------------------
__global__ void k_stats(const float* __restrict__ conv_out, const float* __restrict__ sdw,
                        float* __restrict__ stats) {
    int c = blockIdx.x;                       // 0..255
    const float* src = (c < OCH) ? conv_out : sdw;
    int ch = c & (OCH - 1);
    float sum = 0.f, sq = 0.f;
    for (int i = threadIdx.x; i < NPIX; i += blockDim.x) {
        int b = i >> 10, p = i & 1023;
        float v = src[((size_t)b * OCH + ch) * PIX + p];
        sum += v; sq += v * v;
    }
    __shared__ float ss[256], s2[256];
    ss[threadIdx.x] = sum; s2[threadIdx.x] = sq;
    for (int st = 128; st > 0; st >>= 1) {
        __syncthreads();
        if ((int)threadIdx.x < st) {
            ss[threadIdx.x] += ss[threadIdx.x + st];
            s2[threadIdx.x] += s2[threadIdx.x + st];
        }
    }
    if (threadIdx.x == 0) {
        float m = ss[0] / (float)NPIX;
        float v = s2[0] / (float)NPIX - m * m;
        stats[2 * c]     = m;
        stats[2 * c + 1] = rsqrtf(v + 1e-5f);
    }
}

// ---------------------------------------------------------------------------
// K6: out[:, :128] = relu(BN(conv) - relu(BN(sdw))); out[:, 128:133] = x[:, 64:69]
// ---------------------------------------------------------------------------
__global__ void k_final(const float* __restrict__ conv_out, const float* __restrict__ sdw,
                        const float* __restrict__ stats, const float* __restrict__ x,
                        float* __restrict__ out) {
    int idx = blockIdx.x * blockDim.x + threadIdx.x;
    const int TOT = BB * (OCH + SHC) * PIX;
    if (idx >= TOT) return;
    int p = idx & 1023;
    int t = idx >> 10;
    int c = t % (OCH + SHC);
    int b = t / (OCH + SHC);
    float val;
    if (c < OCH) {
        size_t i = ((size_t)b * OCH + c) * PIX + p;
        float cbn = (conv_out[i] - stats[2 * c]) * stats[2 * c + 1];
        float sbn = (sdw[i] - stats[2 * (OCH + c)]) * stats[2 * (OCH + c) + 1];
        float mux = fmaxf(sbn, 0.f);
        val = fmaxf(cbn - mux, 0.f);
    } else {
        val = x[((size_t)b * XC + CIN + (c - OCH)) * PIX + p];
    }
    out[idx] = val;
}

// ---------------------------------------------------------------------------
extern "C" void kernel_launch(void* const* d_in, const int* in_sizes, int n_in,
                              void* d_out, int out_size, void* d_ws, size_t ws_size,
                              hipStream_t stream) {
    const float* x      = (const float*)d_in[0];
    const float* conv_w = (const float*)d_in[1];
    const float* conv_b = (const float*)d_in[2];
    const float* sk     = (const float*)d_in[3];
    float* out = (float*)d_out;
    (void)in_sizes; (void)n_in; (void)out_size; (void)ws_size;

    char* ws = (char*)d_ws;
    size_t off = 0;
    auto take = [&](size_t bytes) -> char* {
        char* ptr = ws + off;
        off += (bytes + 255) & ~(size_t)255;
        return ptr;
    };
    _Float16* featp  = (_Float16*)take((size_t)BB * CIN * CHW * 2);       // 2.37 MB
    float*    shpp   = (float*)   take((size_t)BB * SHC * CHW * 4);       // 0.37 MB
    _Float16* apack  = (_Float16*)take((size_t)OT * NS * 32 * 16 * 2);    // 0.15 MB
    float*    convo  = (float*)   take((size_t)BB * OCH * PIX * 4);       // 8.39 MB
    float*    sdwb   = (float*)   take((size_t)BB * OCH * PIX * 4);       // 8.39 MB
    float*    stats  = (float*)   take(512 * 4);                          // ~20 MB total

    k_pack_w<<<(OT * NS * 32 * 16 + 255) / 256, 256, 0, stream>>>(conv_w, apack);
    int npad = BB * CIN * CHW + BB * SHC * CHW;
    k_pad<<<(npad + 255) / 256, 256, 0, stream>>>(x, featp, shpp);
    k_conv_wmma<<<NT, 32, 0, stream>>>(featp, apack, conv_b, convo);
    k_sdw<<<(NPIX + 255) / 256, 256, 0, stream>>>(shpp, sk, sdwb);
    k_stats<<<256, 256, 0, stream>>>(convo, sdwb, stats);
    k_final<<<(BB * (OCH + SHC) * PIX + 255) / 256, 256, 0, stream>>>(convo, sdwb, stats, x, out);
}
